// SpatialGCN_40149354283013
// MI455X (gfx1250) — matrix operand order repaired and checked by reading, not compiled, not run on hardware
//
#include <hip/hip_runtime.h>
#include <hip/hip_bf16.h>
#include <math.h>

#define B_    32
#define CIN_  128
#define COUT_ 256
#define KK_   3
#define T_    256
#define V_    25
#define EPS_  1e-5f

typedef __attribute__((ext_vector_type(16))) _Float16 v16h;
typedef __attribute__((ext_vector_type(8)))  _Float16 v8h;
typedef __attribute__((ext_vector_type(8)))  float    v8f;

// ---------------- kernel 1: x_mean[b*CIN+ci] = mean over T*V ----------------
__global__ void xmean_kernel(const float* __restrict__ x, float* __restrict__ xmean) {
    __shared__ float red[256];
    const int bc = blockIdx.x;                     // b*CIN + ci
    const float* p = x + (size_t)bc * (T_ * V_);
    float s = 0.f;
    for (int i = threadIdx.x; i < T_ * V_; i += 256) s += p[i];
    red[threadIdx.x] = s;
    __syncthreads();
    for (int off = 128; off > 0; off >>= 1) {
        if (threadIdx.x < off) red[threadIdx.x] += red[threadIdx.x + off];
        __syncthreads();
    }
    if (threadIdx.x == 0) xmean[bc] = red[0] * (1.0f / (T_ * V_));
}

// ---------------- kernel 2: att[b][k] = softmax_k( (W_att W_gcn) x_mean ) ----
__global__ void att_kernel(const float* __restrict__ Wg, const float* __restrict__ Wa,
                           const float* __restrict__ xmean, float* __restrict__ att) {
    __shared__ float M[KK_ * CIN_];
    __shared__ float lg[B_ * KK_];
    const int tid = threadIdx.x;  // 128 threads
    for (int e = tid; e < KK_ * CIN_; e += 128) {
        const int k = e / CIN_, ci = e % CIN_;
        float s = 0.f;
        for (int o = 0; o < KK_ * COUT_; ++o)
            s = fmaf(Wa[k * (KK_ * COUT_) + o], Wg[o * CIN_ + ci], s);
        M[e] = s;
    }
    __syncthreads();
    for (int e = tid; e < B_ * KK_; e += 128) {
        const int b = e / KK_, k = e % KK_;
        float s = 0.f;
        for (int ci = 0; ci < CIN_; ++ci)
            s = fmaf(M[k * CIN_ + ci], xmean[b * CIN_ + ci], s);
        lg[e] = s;
    }
    __syncthreads();
    if (tid < B_) {
        const float a0 = lg[tid * 3], a1 = lg[tid * 3 + 1], a2 = lg[tid * 3 + 2];
        const float m  = fmaxf(a0, fmaxf(a1, a2));
        const float e0 = __expf(a0 - m), e1 = __expf(a1 - m), e2 = __expf(a2 - m);
        const float inv = 1.0f / (e0 + e1 + e2);
        att[tid * 3 + 0] = e0 * inv;
        att[tid * 3 + 1] = e1 * inv;
        att[tid * 3 + 2] = e2 * inv;
    }
}

// ---------------- kernel 3: fused xa-build + WMMA GEMM + BN epilogue --------
#define MBLK 128                 // output channels per block (8 M-tiles)
#define TC   8                   // timesteps per block
#define NCOL (TC * V_)           // 200 valid columns
#define NPAD 208                 // 13 N-tiles of 16
#define KDIM (KK_ * CIN_)        // 384 = 12 k-steps of 32
#define KSTEPS (KDIM / 32)       // 12
#define KPAD 392                 // row stride (halves); 392*2=784 B, 16B aligned

#define AF_BYTES (MBLK * KPAD * 2)       // 100352
#define BF_OFF   AF_BYTES
#define BF_BYTES (NPAD * KPAD * 2)       // 163072
#define AE_OFF   (BF_OFF + BF_BYTES)     // 263424
#define AE_BYTES 7504                    // 3*25*25 floats, padded
#define AT_OFF   (AE_OFF + AE_BYTES)     // 270928
#define SMEM_TOTAL (AT_OFF + 16)         // 270944 bytes (< 320KB WGP LDS)

// scheduling-group masks (AMDGPU backend): 0x0008 = MFMA/WMMA, 0x0100 = DS-read
#if defined(__has_builtin)
#if __has_builtin(__builtin_amdgcn_sched_group_barrier)
#define SGB(mask, size) __builtin_amdgcn_sched_group_barrier((mask), (size), 0)
#endif
#endif
#ifndef SGB
#define SGB(mask, size)
#endif

__global__ void __launch_bounds__(256, 1)
spatial_gcn_main(const float* __restrict__ x, const float* __restrict__ Wg,
                 const float* __restrict__ A, const float* __restrict__ edge,
                 const float* __restrict__ att,
                 const float* __restrict__ gamma, const float* __restrict__ beta,
                 const float* __restrict__ rmean, const float* __restrict__ rvar,
                 float* __restrict__ out) {
    extern __shared__ char smem[];
    _Float16* Af = (_Float16*)(smem);            // W_flat tile  [MBLK][KPAD]
    _Float16* Bf = (_Float16*)(smem + BF_OFF);   // xa^T         [NPAD][KPAD]
    float*    Ae = (float*)(smem + AE_OFF);      // A*edge       [3][25][25]
    float*    atb = (float*)(smem + AT_OFF);     // att row      [3]

    const int tid = threadIdx.x;
    const int c0 = blockIdx.x * MBLK;
    const int t0 = blockIdx.y * TC;
    const int b  = blockIdx.z;

    // ---- load A_eff and att[b], zero Bf padding rows ----
    for (int i = tid; i < KK_ * V_ * V_; i += 256) Ae[i] = A[i] * edge[i];
    if (tid < KK_) atb[tid] = att[b * KK_ + tid];
    for (int i = tid; i < (NPAD - NCOL) * KPAD; i += 256) {
        const int r = NCOL + i / KPAD, cc = i % KPAD;
        Bf[r * KPAD + cc] = (_Float16)0.f;
    }
    __syncthreads();

    // ---- stage 1: Bf[(t*25+w)][k*128+ci] = att_k * sum_v x[ci,t,v]*Ae[k,v,w]
    for (int j = 0; j < (KK_ * CIN_ * TC) / 256; ++j) {   // 12 iterations
        const int idx = tid + 256 * j;
        const int k   = idx / (CIN_ * TC);
        const int rem = idx % (CIN_ * TC);
        const int ci  = rem / TC;
        const int t   = rem % TC;
        const float* xr = x + (size_t)((b * CIN_ + ci) * T_ + (t0 + t)) * V_;
        __builtin_prefetch(xr + V_, 0, 1);                // global_prefetch_b8
        float xv[V_];
        #pragma unroll
        for (int v = 0; v < V_; ++v) xv[v] = xr[v];
        const float ak = atb[k];
        const float* Aek = Ae + k * V_ * V_;
        const int kkcol = k * CIN_ + ci;
        #pragma unroll
        for (int w = 0; w < V_; ++w) {
            float s = 0.f;
            #pragma unroll
            for (int v = 0; v < V_; ++v) s = fmaf(xv[v], Aek[v * V_ + w], s);
            Bf[(t * V_ + w) * KPAD + kkcol] = (_Float16)(ak * s);
        }
    }
    // ---- stage 1b: Af[m][k*128+ci] = W_gcn[(k*COUT + c0+m)*CIN + ci] ----
    for (int j = 0; j < (MBLK * KDIM) / 256; ++j) {       // 192 iterations
        const int idx = tid + 256 * j;
        const int m  = idx / KDIM;
        const int kk = idx % KDIM;
        const int k  = kk / CIN_, ci = kk % CIN_;
        Af[m * KPAD + kk] = (_Float16)Wg[((size_t)(k * COUT_) + c0 + m) * CIN_ + ci];
    }
    __syncthreads();

    // ---- stage 2: WMMA.  wave wv owns M-tile wv; iterates 13 N-tiles ----
    const int wv   = tid >> 5;       // 0..7
    const int lane = tid & 31;
    const int lmod = lane & 15;
    const int lhi  = lane >> 4;      // 0/1

    // A fragments: 16-bit A 16x32 layout -> halves 0-7: K = kb*32 + lhi*8 + 0..7,
    // halves 8-15: K = kb*32 + 16 + lhi*8 + 0..7  (two ds_load_b128 per fragment)
    v16h afr[KSTEPS];
    {
        const _Float16* arow = Af + (wv * 16 + lmod) * KPAD;
        #pragma unroll
        for (int kb = 0; kb < KSTEPS; ++kb) {
            const int ka = kb * 32 + lhi * 8;
            v8h lo = *(const v8h*)(arow + ka);
            v8h hi = *(const v8h*)(arow + ka + 16);
            afr[kb] = __builtin_shufflevector(lo, hi, 0,1,2,3,4,5,6,7,8,9,10,11,12,13,14,15);
        }
    }
    // BN constants for this wave's 8 accumulator rows (fixed across N-tiles)
    float scl[8], shf[8];
    #pragma unroll
    for (int g = 0; g < 8; ++g) {
        const int c = c0 + wv * 16 + g + lhi * 8;
        const float inv = rsqrtf(rvar[c] + EPS_);
        const float sc  = gamma[c] * inv;
        scl[g] = sc;
        shf[g] = beta[c] - rmean[c] * sc;
    }

    for (int nt = 0; nt < NPAD / 16; ++nt) {              // 13 N-tiles
        v8f acc = {};
        const _Float16* brow = Bf + (nt * 16 + lmod) * KPAD;

        #pragma unroll
        for (int kb = 0; kb < KSTEPS; ++kb) {
            const int kbase = kb * 32 + lhi * 8;
            v8h p0 = *(const v8h*)(brow + kbase);
            v8h p1 = *(const v8h*)(brow + kbase + 16);
            v16h bf = __builtin_shufflevector(p0, p1, 0,1,2,3,4,5,6,7,8,9,10,11,12,13,14,15);
            acc = __builtin_amdgcn_wmma_f32_16x16x32_f16(
                false, afr[kb], false, bf, (short)0, acc, false, false);
        }

        // Force a software-pipelined schedule for this (single) basic block:
        //   prologue: 6 ds_reads (3 fragments in flight)
        //   steady:   9 x (1 WMMA, 2 ds_reads)
        //   drain:    3 x WMMA
        // -> waitcnt pass emits partial s_wait_dscnt instead of 0x0 per WMMA.
        SGB(0x0100, 6);
        #pragma unroll
        for (int i = 0; i < 9; ++i) {
            SGB(0x0008, 1);
            SGB(0x0100, 2);
        }
        #pragma unroll
        for (int i = 0; i < 3; ++i) {
            SGB(0x0008, 1);
        }

        const int col = nt * 16 + lmod;
        if (col < NCOL) {
            const int t = t0 + col / V_;
            const int w = col % V_;
            #pragma unroll
            for (int g = 0; g < 8; ++g) {
                const int c = c0 + wv * 16 + g + lhi * 8;
                out[(size_t)((b * COUT_ + c) * T_ + t) * V_ + w] = acc[g] * scl[g] + shf[g];
            }
        }
    }
}

// ---------------- host-side launcher ----------------------------------------
extern "C" void kernel_launch(void* const* d_in, const int* in_sizes, int n_in,
                              void* d_out, int out_size, void* d_ws, size_t ws_size,
                              hipStream_t stream) {
    const float* x     = (const float*)d_in[0];
    const float* Wg    = (const float*)d_in[1];
    const float* A     = (const float*)d_in[2];
    const float* edge  = (const float*)d_in[3];
    const float* Wa    = (const float*)d_in[4];
    const float* gamma = (const float*)d_in[5];
    const float* beta  = (const float*)d_in[6];
    const float* rmean = (const float*)d_in[7];
    const float* rvar  = (const float*)d_in[8];
    float* out = (float*)d_out;

    float* xmean = (float*)d_ws;          // B*CIN floats
    float* att   = xmean + B_ * CIN_;     // B*K floats

    xmean_kernel<<<B_ * CIN_, 256, 0, stream>>>(x, xmean);
    att_kernel<<<1, 128, 0, stream>>>(Wg, Wa, xmean, att);

    dim3 grid(COUT_ / MBLK, T_ / TC, B_);
    spatial_gcn_main<<<grid, 256, SMEM_TOTAL, stream>>>(
        x, Wg, A, edge, att, gamma, beta, rmean, rvar, out);
}
